// DGSL_3453153706625
// MI455X (gfx1250) — matrix operand order repaired and checked by reading, not compiled, not run on hardware
//
#include <hip/hip_runtime.h>
#include <hip/hip_bf16.h>

#define N_MICRO_ 131072
#define E_MICRO_ 1048576
#define N_MACRO_ 6400
#define E_MACRO_ 51200
#define BB 64
#define TT 50
#define NGG 5
#define IN_DIM_ 384
#define HH 256
#define SS 64
#define NPM 100   // nodes per macro

typedef __attribute__((ext_vector_type(16))) __bf16 bf16x16;
typedef __attribute__((ext_vector_type(8)))  float  f32x8;

union BF16Vec {
    bf16x16 v;
    unsigned short u[16];
    uint4 q[2];
};

__device__ __forceinline__ unsigned short f2bf(float f) {
    unsigned int x = __float_as_uint(f);
    unsigned int r = x + 0x7FFFu + ((x >> 16) & 1u);   // RNE
    return (unsigned short)(r >> 16);
}

__device__ __forceinline__ unsigned int pack2bf(float lo, float hi) {
    return (unsigned int)f2bf(lo) | ((unsigned int)f2bf(hi) << 16);
}

// ---------------------------------------------------------------------------
// Generic bf16 WMMA GEMM, 4-wave workgroup sharing the A tile through LDS.
//   C[M,N] = act( A[M,K](f32) * Bt[N_pad,K](bf16)^T + bias + res )
// Block = 128 threads (4 waves): wave w covers N columns [blockIdx.y*256+w*64, +64).
// A tile (16 rows x 32 K, f32) is cooperatively loaded/converted to bf16 in LDS
// once per K-step and re-read by every wave (ds_load_b128).
// Optional fused second output (GCN self-term):
//   C2[m,n] = raw[m,n] * rowscale[m]^2 + bias2[n]
// M % 16 == 0, K % 32 == 0, grid.y = N_pad/256 (N_pad % 256 == 0).
// ---------------------------------------------------------------------------
__global__ __launch_bounds__(128)
void wmma_gemm_bf16(const float* __restrict__ A, const unsigned short* __restrict__ Bt,
                    const float* __restrict__ bias, const float* __restrict__ res,
                    float* __restrict__ C, int K, int Nvalid, int ldc, int act,
                    const float* __restrict__ rowscale, float* __restrict__ C2,
                    const float* __restrict__ bias2)
{
    __shared__ unsigned short sA[16 * 32];   // 1 KB bf16 A tile

    const int tid  = threadIdx.x;
    const int wave = tid >> 5;
    const int lane = tid & 31;
    const int half = lane >> 4;   // 0 or 1
    const int l16  = lane & 15;
    const int m0 = blockIdx.x * 16;
    const int n0 = blockIdx.y * 256 + wave * 64;

    f32x8 c0 = {}, c1 = {}, c2 = {}, c3 = {};

    const unsigned short* b0p = Bt + (size_t)(n0 +  0 + l16) * K;
    const unsigned short* b1p = Bt + (size_t)(n0 + 16 + l16) * K;
    const unsigned short* b2p = Bt + (size_t)(n0 + 32 + l16) * K;
    const unsigned short* b3p = Bt + (size_t)(n0 + 48 + l16) * K;

    // cooperative A loader mapping: 128 threads cover 16 rows x 8 col-groups of 4
    const int arow_i = tid >> 3;          // 0..15
    const int acol   = (tid & 7) * 4;     // 0,4,..,28
    const float* aptr = A + (size_t)(m0 + arow_i) * K + acol;
    unsigned short* sdst = sA + arow_i * 32 + acol;

    for (int kk = 0; kk < K; kk += 32) {
        float4 a = *(const float4*)(aptr + kk);
        if (kk + 32 < K) {                               // uniform branch
            __builtin_prefetch(aptr + kk + 32, 0, 1);    // global_prefetch_b8
        }
        __syncthreads();                                 // prior LDS reads done
        *(uint2*)sdst = make_uint2(pack2bf(a.x, a.y), pack2bf(a.z, a.w));
        __syncthreads();                                 // tile visible to all waves

        // A fragment: two ds_load_b128 per lane (K chunks [16*half..+7], [16*half+8..+7])
        BF16Vec av;
        av.q[0] = *(const uint4*)(sA + l16 * 32 + 16 * half);
        av.q[1] = *(const uint4*)(sA + l16 * 32 + 16 * half + 8);

        // B fragments: lane holds col n, 16 K-contiguous bf16 = 2 x global b128
        const int kb = kk + 16 * half;
        BF16Vec bv0, bv1, bv2, bv3;
        bv0.q[0] = *(const uint4*)(b0p + kb); bv0.q[1] = *(const uint4*)(b0p + kb + 8);
        bv1.q[0] = *(const uint4*)(b1p + kb); bv1.q[1] = *(const uint4*)(b1p + kb + 8);
        bv2.q[0] = *(const uint4*)(b2p + kb); bv2.q[1] = *(const uint4*)(b2p + kb + 8);
        bv3.q[0] = *(const uint4*)(b3p + kb); bv3.q[1] = *(const uint4*)(b3p + kb + 8);

        c0 = __builtin_amdgcn_wmma_f32_16x16x32_bf16(false, av.v, false, bv0.v, (short)0, c0, false, false);
        c1 = __builtin_amdgcn_wmma_f32_16x16x32_bf16(false, av.v, false, bv1.v, (short)0, c1, false, false);
        c2 = __builtin_amdgcn_wmma_f32_16x16x32_bf16(false, av.v, false, bv2.v, (short)0, c2, false, false);
        c3 = __builtin_amdgcn_wmma_f32_16x16x32_bf16(false, av.v, false, bv3.v, (short)0, c3, false, false);
    }

    // ---- store: C/D layout: VGPR v -> M = v + 8*half ; lane -> N = l16
    #pragma unroll
    for (int j = 0; j < 4; ++j) {
        f32x8 cc = (j == 0) ? c0 : (j == 1) ? c1 : (j == 2) ? c2 : c3;
        const int n = n0 + j * 16 + l16;
        if (n < Nvalid) {
            const float bs = bias ? bias[n] : 0.0f;
            #pragma unroll
            for (int v = 0; v < 8; ++v) {
                const int m = m0 + v + 8 * half;
                float raw = cc[v];
                float val = raw + bs;
                if (res) val += res[(size_t)m * ldc + n];
                if (act) val = fmaxf(val, 0.0f);
                C[(size_t)m * ldc + n] = val;
                if (C2) {
                    float di = rowscale[m];
                    C2[(size_t)m * ldc + n] = raw * di * di + bias2[n];
                }
            }
        }
    }
}

// ---------------------------------------------------------------------------
// Weight prep: Wt[n][k] = bf16(W[k][n]) for n < N, zero for padding rows.
// grid.x = N_pad
// ---------------------------------------------------------------------------
__global__ void convert_wt(const float* __restrict__ W, unsigned short* __restrict__ Wt,
                           int K, int N)
{
    const int n = blockIdx.x;
    for (int k = threadIdx.x; k < K; k += blockDim.x)
        Wt[(size_t)n * K + k] = (n < N) ? f2bf(W[(size_t)k * N + n]) : (unsigned short)0;
}

// ---------------------------------------------------------------------------
// GCN helpers
// ---------------------------------------------------------------------------
__global__ void fill1(float* p, int n) {
    int i = blockIdx.x * blockDim.x + threadIdx.x;
    if (i < n) p[i] = 1.0f;
}

__global__ void deg_acc(float* __restrict__ deg, const int* __restrict__ dst,
                        const float* __restrict__ ew, int E)
{
    int i = blockIdx.x * blockDim.x + threadIdx.x;
    if (i < E) atomicAdd(&deg[dst[i]], ew[i]);
}

__global__ void rsqrt_k(float* p, int n) {
    int i = blockIdx.x * blockDim.x + threadIdx.x;
    if (i < n) p[i] = rsqrtf(p[i]);
}

// 4 edges per 256-thread block; each lane handles 4 channels via float4 load.
__global__ __launch_bounds__(256)
void edge_scatter(float* __restrict__ out, const float* __restrict__ h,
                  const int* __restrict__ src, const int* __restrict__ dst,
                  const float* __restrict__ ew, const float* __restrict__ dinv)
{
    const int e  = blockIdx.x * 4 + (threadIdx.x >> 6);
    const int c  = (threadIdx.x & 63) * 4;
    const int s  = src[e];
    const int d  = dst[e];
    const float norm = dinv[s] * ew[e] * dinv[d];
    float4 hv = *(const float4*)(h + (size_t)s * HH + c);
    float* op = out + (size_t)d * HH + c;
    atomicAdd(op + 0, hv.x * norm);
    atomicAdd(op + 1, hv.y * norm);
    atomicAdd(op + 2, hv.z * norm);
    atomicAdd(op + 3, hv.w * norm);
}

// ---------------------------------------------------------------------------
// Gather + mean over NG neighbors, masked
// ---------------------------------------------------------------------------
__global__ __launch_bounds__(256)
void gather_mean(const float* __restrict__ gout, const int* __restrict__ idx,
                 const unsigned char* __restrict__ mask, float* __restrict__ seq)
{
    const int bt = blockIdx.x;
    const int c = threadIdx.x;
    float acc = 0.0f;
    #pragma unroll
    for (int g = 0; g < NGG; ++g) {
        int node = idx[bt * NGG + g];
        acc += gout[(size_t)node * HH + c];
    }
    acc *= (1.0f / NGG);
    if (!mask[bt]) acc = 0.0f;
    seq[(size_t)bt * HH + c] = acc;
}

// ---------------------------------------------------------------------------
// Mamba selective scan: one block per batch, one thread per channel h.
// State [SS] lives in registers; B_t/C_t staged in LDS.
// ---------------------------------------------------------------------------
__global__ __launch_bounds__(256)
void mamba_scan(const float* __restrict__ xz, const float* __restrict__ dbc,
                const float* __restrict__ dt_bias, const float* __restrict__ A_log,
                const float* __restrict__ Dp, float* __restrict__ g)
{
    const int b = blockIdx.x;
    const int h = threadIdx.x;
    __shared__ float sB[SS];
    __shared__ float sC[SS];

    float hst[SS];
    #pragma unroll
    for (int n = 0; n < SS; ++n) hst[n] = 0.0f;

    const float Ah  = -expf(A_log[h]);
    const float dph = Dp[h];
    const float dtb = dt_bias[0];

    for (int t = 0; t < TT; ++t) {
        const size_t bt = (size_t)b * TT + t;
        const float* drow = dbc + bt * 192;
        if (h < SS)          sB[h]      = drow[1 + h];
        else if (h < 2 * SS) sC[h - SS] = drow[1 + SS + (h - SS)];
        __syncthreads();

        float dtr = drow[0] + dtb;
        float dt  = (dtr > 20.0f) ? dtr : log1pf(expf(dtr));   // softplus
        float x = xz[bt * (2 * HH) + h];
        float z = xz[bt * (2 * HH) + HH + h];
        float decay = expf(dt * Ah);
        float dx = dt * x;

        float y = 0.0f;
        #pragma unroll
        for (int n = 0; n < SS; ++n) {
            hst[n] = decay * hst[n] + dx * sB[n];
            y += hst[n] * sC[n];
        }
        y += dph * x;
        float sz = z / (1.0f + expf(-z));   // silu
        g[bt * HH + h] = y * sz;
        __syncthreads();
    }
}

// ---------------------------------------------------------------------------
// Pools
// ---------------------------------------------------------------------------
__global__ __launch_bounds__(256)
void macro_pool(const float* __restrict__ feat, float* __restrict__ pooled)
{
    const int b = blockIdx.x, h = threadIdx.x;
    float acc = 0.0f;
    for (int i = 0; i < NPM; ++i)
        acc += feat[((size_t)b * NPM + i) * HH + h];
    pooled[(size_t)b * (2 * HH) + h] = acc * (1.0f / NPM);
}

__global__ __launch_bounds__(256)
void micro_pick(const float* __restrict__ seqo, float* __restrict__ pooled)
{
    const int b = blockIdx.x, h = threadIdx.x;
    pooled[(size_t)b * (2 * HH) + HH + h] = seqo[((size_t)b * TT + (TT - 1)) * HH + h];
}

// ---------------------------------------------------------------------------
extern "C" void kernel_launch(void* const* d_in, const int* in_sizes, int n_in,
                              void* d_out, int out_size, void* d_ws, size_t ws_size,
                              hipStream_t stream)
{
    const float* micro_x  = (const float*)d_in[0];
    const float* micro_ew = (const float*)d_in[1];
    const float* macro_x  = (const float*)d_in[2];
    const float* macro_ew = (const float*)d_in[3];
    const float* Wg_micro = (const float*)d_in[4];
    const float* bg_micro = (const float*)d_in[5];
    const float* Wg_macro = (const float*)d_in[6];
    const float* bg_macro = (const float*)d_in[7];
    const float* W_in     = (const float*)d_in[8];
    const float* W_dtBC   = (const float*)d_in[9];
    const float* dt_bias  = (const float*)d_in[10];
    const float* A_log    = (const float*)d_in[11];
    const float* Dp       = (const float*)d_in[12];
    const float* W_out    = (const float*)d_in[13];
    const float* W1       = (const float*)d_in[14];
    const float* b1       = (const float*)d_in[15];
    const float* W2       = (const float*)d_in[16];
    const float* b2       = (const float*)d_in[17];
    const int*   micro_ei = (const int*)d_in[18];
    const int*   gidx     = (const int*)d_in[19];
    const unsigned char* mask = (const unsigned char*)d_in[20];
    const int*   macro_ei = (const int*)d_in[21];

    char* ws = (char*)d_ws;
    size_t off = 0;
    auto alloc = [&](size_t bytes) -> void* {
        void* p = ws + off;
        off += (bytes + 255) & ~(size_t)255;
        return p;
    };

    // bf16 transposed weights (rows padded to N_pad, multiple of 256)
    unsigned short* wt_gmi  = (unsigned short*)alloc((size_t)256 * 384 * 2);
    unsigned short* wt_gma  = (unsigned short*)alloc((size_t)256 * 384 * 2);
    unsigned short* wt_in   = (unsigned short*)alloc((size_t)512 * 256 * 2);
    unsigned short* wt_dtbc = (unsigned short*)alloc((size_t)256 * 256 * 2);
    unsigned short* wt_out  = (unsigned short*)alloc((size_t)256 * 256 * 2);
    unsigned short* wt_1    = (unsigned short*)alloc((size_t)256 * 512 * 2);
    unsigned short* wt_2    = (unsigned short*)alloc((size_t)512 * 256 * 2);
    // graph scratch
    float* deg_mi = (float*)alloc((size_t)N_MICRO_ * 4);
    float* h_mi   = (float*)alloc((size_t)N_MICRO_ * HH * 4);
    float* o_mi   = (float*)alloc((size_t)N_MICRO_ * HH * 4);
    float* deg_ma = (float*)alloc((size_t)N_MACRO_ * 4);
    float* h_ma   = (float*)alloc((size_t)N_MACRO_ * HH * 4);
    float* o_ma   = (float*)alloc((size_t)N_MACRO_ * HH * 4);
    // sequence scratch
    float* seq  = (float*)alloc((size_t)BB * TT * HH * 4);
    float* xz   = (float*)alloc((size_t)BB * TT * 2 * HH * 4);
    float* dbc  = (float*)alloc((size_t)BB * TT * 192 * 4);
    float* gbuf = (float*)alloc((size_t)BB * TT * HH * 4);
    float* seqo = (float*)alloc((size_t)BB * TT * HH * 4);
    float* pooled = (float*)alloc((size_t)BB * 2 * HH * 4);
    float* hdn    = (float*)alloc((size_t)BB * HH * 4);

    // ---- weight prep (transpose + f32->bf16) ----
    convert_wt<<<256, 256, 0, stream>>>(Wg_micro, wt_gmi, 384, 256);
    convert_wt<<<256, 256, 0, stream>>>(Wg_macro, wt_gma, 384, 256);
    convert_wt<<<512, 256, 0, stream>>>(W_in,     wt_in,  256, 512);
    convert_wt<<<256, 256, 0, stream>>>(W_dtBC,   wt_dtbc,256, 129);
    convert_wt<<<256, 256, 0, stream>>>(W_out,    wt_out, 256, 256);
    convert_wt<<<256, 256, 0, stream>>>(W1,       wt_1,   512, 256);
    convert_wt<<<512, 256, 0, stream>>>(W2,       wt_2,   256, 512);

    // ---- micro GCN ----
    fill1<<<(N_MICRO_ + 255) / 256, 256, 0, stream>>>(deg_mi, N_MICRO_);
    deg_acc<<<(E_MICRO_ + 255) / 256, 256, 0, stream>>>(deg_mi, micro_ei + E_MICRO_, micro_ew, E_MICRO_);
    rsqrt_k<<<(N_MICRO_ + 255) / 256, 256, 0, stream>>>(deg_mi, N_MICRO_);
    // h = x@W  and fused  o = h*dinv^2 + b
    wmma_gemm_bf16<<<dim3(N_MICRO_ / 16, 1), 128, 0, stream>>>(
        micro_x, wt_gmi, nullptr, nullptr, h_mi, 384, 256, 256, 0,
        deg_mi, o_mi, bg_micro);
    edge_scatter<<<E_MICRO_ / 4, 256, 0, stream>>>(o_mi, h_mi, micro_ei, micro_ei + E_MICRO_, micro_ew, deg_mi);

    // ---- macro GCN ----
    fill1<<<(N_MACRO_ + 255) / 256, 256, 0, stream>>>(deg_ma, N_MACRO_);
    deg_acc<<<(E_MACRO_ + 255) / 256, 256, 0, stream>>>(deg_ma, macro_ei + E_MACRO_, macro_ew, E_MACRO_);
    rsqrt_k<<<(N_MACRO_ + 255) / 256, 256, 0, stream>>>(deg_ma, N_MACRO_);
    wmma_gemm_bf16<<<dim3(N_MACRO_ / 16, 1), 128, 0, stream>>>(
        macro_x, wt_gma, nullptr, nullptr, h_ma, 384, 256, 256, 0,
        deg_ma, o_ma, bg_macro);
    edge_scatter<<<E_MACRO_ / 4, 256, 0, stream>>>(o_ma, h_ma, macro_ei, macro_ei + E_MACRO_, macro_ew, deg_ma);

    // ---- gather + mean -> seq [B*T, 256] ----
    gather_mean<<<BB * TT, 256, 0, stream>>>(o_mi, gidx, mask, seq);

    // ---- mamba block ----
    const int MT = BB * TT;  // 3200
    wmma_gemm_bf16<<<dim3(MT / 16, 2), 128, 0, stream>>>(
        seq, wt_in, nullptr, nullptr, xz, 256, 512, 512, 0, nullptr, nullptr, nullptr);
    wmma_gemm_bf16<<<dim3(MT / 16, 1), 128, 0, stream>>>(
        seq, wt_dtbc, nullptr, nullptr, dbc, 256, 129, 192, 0, nullptr, nullptr, nullptr);
    mamba_scan<<<BB, 256, 0, stream>>>(xz, dbc, dt_bias, A_log, Dp, gbuf);
    wmma_gemm_bf16<<<dim3(MT / 16, 1), 128, 0, stream>>>(
        gbuf, wt_out, nullptr, seq, seqo, 256, 256, 256, 0, nullptr, nullptr, nullptr);  // + residual u

    // ---- pooling + concat ----
    macro_pool<<<BB, 256, 0, stream>>>(o_ma, pooled);
    micro_pick<<<BB, 256, 0, stream>>>(seqo, pooled);

    // ---- final MLP ----
    wmma_gemm_bf16<<<dim3(BB / 16, 1), 128, 0, stream>>>(
        pooled, wt_1, b1, nullptr, hdn, 512, 256, 256, 1, nullptr, nullptr, nullptr);    // relu
    wmma_gemm_bf16<<<dim3(BB / 16, 2), 128, 0, stream>>>(
        hdn, wt_2, b2, nullptr, (float*)d_out, 256, 512, 512, 0, nullptr, nullptr, nullptr);
}
